// StatPooling_45268955300240
// MI455X (gfx1250) — matrix operand order repaired and checked by reading, not compiled, not run on hardware
//
#include <hip/hip_runtime.h>

typedef __attribute__((ext_vector_type(16))) __bf16 v16bf;
typedef __attribute__((ext_vector_type(8)))  __bf16 v8bf;
typedef __attribute__((ext_vector_type(8)))  float  v8f;

#define TILE_M 128
#define TILE_N 128
#define TILE_K 32
#define LDSR   40          // padded LDS row stride (bf16 elements) -> 80B, 16B-aligned chunks
#define NPIX   4608        // 32*12*12

__device__ inline unsigned short bf16_rne(float x) {
  unsigned u = __float_as_uint(x);
  unsigned r = u + 0x7FFFu + ((u >> 16) & 1u);
  return (unsigned short)(r >> 16);
}

union Frag { v16bf v; v8bf h[2]; };
union F4U  { float4 v; float f[4]; };

// C[N,M] = (A*scale+shift)[N,K] x W[K,M] + cbias, A gathered from up to 3 concat sources.
__global__ __launch_bounds__(256) void gemm_bn_split(
    const float* __restrict__ A0, const float* __restrict__ A1, const float* __restrict__ A2,
    int K0, int K1, int lda0, int lda1, int lda2,
    const float* __restrict__ W, int M, int K,
    const float* __restrict__ scale, const float* __restrict__ shift,
    const float* __restrict__ cbias, float* __restrict__ Out)
{
  __shared__ unsigned short sAh[TILE_M * LDSR];
  __shared__ unsigned short sAl[TILE_M * LDSR];
  __shared__ unsigned short sBh[TILE_N * LDSR];
  __shared__ unsigned short sBl[TILE_N * LDSR];

  const int t = threadIdx.x;
  const int rowBase = blockIdx.x * TILE_M;   // pixel rows
  const int n0 = blockIdx.y * TILE_N;        // output channels

  const int wave = t >> 5;
  const int lane = t & 31;
  const int wm = wave >> 2;                  // 0..1 : 64-row strip
  const int wn = wave & 3;                   // 0..3 : 32-col strip
  const int l15 = lane & 15;
  const int hi  = lane >> 4;

  v8f acc[4][2] = {};

  for (int k0 = 0; k0 < K; k0 += TILE_K) {
    // ---- stage A tile 128x32: BN-fold + bf16 hi/lo split ----
#pragma unroll
    for (int i = 0; i < 4; ++i) {
      int idx = t + i * 256;                 // float4 id 0..1023
      int row = idx >> 3;                    // 0..127
      int c4  = (idx & 7) << 2;              // k-in-tile 0..28 (concat bounds are /4)
      int gk  = k0 + c4;
      const float* src; int col; int ld;
      if (gk < K0)      { src = A0; col = gk;      ld = lda0; }
      else if (gk < K1) { src = A1; col = gk - K0; ld = lda1; }
      else              { src = A2; col = gk - K1; ld = lda2; }
      F4U x;  x.v  = *(const float4*)(src + (size_t)(rowBase + row) * ld + col);
      F4U sc; sc.v = *(const float4*)(scale + gk);
      F4U sf; sf.v = *(const float4*)(shift + gk);
#pragma unroll
      for (int j = 0; j < 4; ++j) {
        float y = x.f[j] * sc.f[j] + sf.f[j];
        unsigned short h = bf16_rne(y);
        unsigned short l = bf16_rne(y - __uint_as_float((unsigned)h << 16));
        sAh[row * LDSR + c4 + j] = h;
        sAl[row * LDSR + c4 + j] = l;
      }
    }
    // ---- stage W tile 32x128, transposed into LDS as [n][k] ----
#pragma unroll
    for (int i = 0; i < 4; ++i) {
      int idx = t + i * 256;
      int kr = idx >> 5;                     // 0..31
      int c4 = (idx & 31) << 2;              // 0..124
      const float* wp = W + (size_t)(k0 + kr) * M + n0 + c4;
      F4U x; x.v = *(const float4*)wp;
      if (k0 + TILE_K < K) __builtin_prefetch(wp + (size_t)TILE_K * M, 0, 0);
#pragma unroll
      for (int j = 0; j < 4; ++j) {
        unsigned short h = bf16_rne(x.f[j]);
        unsigned short l = bf16_rne(x.f[j] - __uint_as_float((unsigned)h << 16));
        sBh[(c4 + j) * LDSR + kr] = h;
        sBl[(c4 + j) * LDSR + kr] = l;
      }
    }
    __syncthreads();

    // B fragments: b[j] = B[k=16*hi+j, n=l15]  -> two contiguous 16B chunks
    Frag bh[2], bl[2];
#pragma unroll
    for (int nt = 0; nt < 2; ++nt) {
      int off = (wn * 32 + nt * 16 + l15) * LDSR + hi * 16;
      bh[nt].h[0] = *(const v8bf*)(const void*)&sBh[off];
      bh[nt].h[1] = *(const v8bf*)(const void*)&sBh[off + 8];
      bl[nt].h[0] = *(const v8bf*)(const void*)&sBl[off];
      bl[nt].h[1] = *(const v8bf*)(const void*)&sBl[off + 8];
    }
    // A fragments: a[j]=A[m, 16*hi+j]  -> two contiguous 16B chunks
#pragma unroll
    for (int mt = 0; mt < 4; ++mt) {
      int off = (wm * 64 + mt * 16 + l15) * LDSR + hi * 16;
      Frag ah, al;
      ah.h[0] = *(const v8bf*)(const void*)&sAh[off];
      ah.h[1] = *(const v8bf*)(const void*)&sAh[off + 8];
      al.h[0] = *(const v8bf*)(const void*)&sAl[off];
      al.h[1] = *(const v8bf*)(const void*)&sAl[off + 8];
#pragma unroll
      for (int nt = 0; nt < 2; ++nt) {
        v8f c = acc[mt][nt];
        c = __builtin_amdgcn_wmma_f32_16x16x32_bf16(false, ah.v, false, bh[nt].v, (short)0, c, false, false);
        c = __builtin_amdgcn_wmma_f32_16x16x32_bf16(false, ah.v, false, bl[nt].v, (short)0, c, false, false);
        c = __builtin_amdgcn_wmma_f32_16x16x32_bf16(false, al.v, false, bh[nt].v, (short)0, c, false, false);
        acc[mt][nt] = c;
      }
    }
    __syncthreads();
  }

  // epilogue: conv bias + fp32 store (C/D layout: vgpr r -> M=r+8*hi, N=l15)
#pragma unroll
  for (int nt = 0; nt < 2; ++nt) {
    int gn = n0 + wn * 32 + nt * 16 + l15;
    float bias = cbias[gn];
#pragma unroll
    for (int mt = 0; mt < 4; ++mt) {
      v8f c = acc[mt][nt];
#pragma unroll
      for (int r = 0; r < 8; ++r) {
        int gm = rowBase + wm * 64 + mt * 16 + r + 8 * hi;
        Out[(size_t)gm * M + gn] = c[r] + bias;
      }
    }
  }
}

__global__ void bnfold_kernel(const float* __restrict__ g, const float* __restrict__ b,
                              const float* __restrict__ m, const float* __restrict__ v,
                              float* __restrict__ scale, float* __restrict__ shift, int C)
{
  int i = blockIdx.x * 256 + threadIdx.x;
  if (i < C) {
    float s = g[i] * rsqrtf(v[i] + 1e-3f);
    scale[i] = s;
    shift[i] = b[i] - m[i] * s;
  }
}

__global__ void gap_kernel(const float* __restrict__ x, float* __restrict__ out)
{
  int c = blockIdx.x * 256 + threadIdx.x;    // 0..2047
  int b = blockIdx.y;                        // 0..31
  const float* p = x + (size_t)b * 144 * 2048 + c;
  float s = 0.f;
  for (int i = 0; i < 144; ++i) s += p[(size_t)i * 2048];
  out[b * 2048 + c] = s * (1.0f / 144.0f);
}

extern "C" void kernel_launch(void* const* d_in, const int* in_sizes, int n_in,
                              void* d_out, int out_size, void* d_ws, size_t ws_size,
                              hipStream_t stream)
{
  (void)in_sizes; (void)n_in; (void)out_size; (void)ws_size;
  const float* stat = (const float*)d_in[0];
  const float* x0   = (const float*)d_in[1];
  const float* g1   = (const float*)d_in[2];
  const float* be1  = (const float*)d_in[3];
  const float* m1   = (const float*)d_in[4];
  const float* v1   = (const float*)d_in[5];
  const float* w1   = (const float*)d_in[6];
  const float* cb1  = (const float*)d_in[7];
  const float* g26  = (const float*)d_in[8];
  const float* be26 = (const float*)d_in[9];
  const float* m26  = (const float*)d_in[10];
  const float* v26  = (const float*)d_in[11];
  const float* w26  = (const float*)d_in[12];
  const float* cb26 = (const float*)d_in[13];
  const float* g7   = (const float*)d_in[14];
  const float* be7  = (const float*)d_in[15];
  const float* m7   = (const float*)d_in[16];
  const float* v7   = (const float*)d_in[17];
  const float* w7   = (const float*)d_in[18];
  const float* cb7  = (const float*)d_in[19];
  const float* g8   = (const float*)d_in[20];
  const float* be8  = (const float*)d_in[21];
  const float* m8   = (const float*)d_in[22];
  const float* v8_  = (const float*)d_in[23];
  const float* w8   = (const float*)d_in[24];
  const float* cb8  = (const float*)d_in[25];

  // workspace layout (~302.5 MB)
  float* xA  = (float*)d_ws;
  float* xB  = xA + (size_t)NPIX * 6144;
  float* x7  = xB + (size_t)NPIX * 6144;
  float* x8  = x7 + (size_t)NPIX * 2048;
  float* scb = x8 + (size_t)NPIX * 2048;     // 8 slots x 8192 scale
  float* shb = scb + 8 * 8192;               // 8 slots x 8192 shift

  dim3 blk(256);

  // fold all BN layers into scale/shift
  bnfold_kernel<<<dim3(3),   blk, 0, stream>>>(g1, be1, m1, v1, scb + 0 * 8192, shb + 0 * 8192, 768);
  for (int l = 0; l < 5; ++l)
    bnfold_kernel<<<dim3(24), blk, 0, stream>>>(g26 + (size_t)l * 6144, be26 + (size_t)l * 6144,
                                                m26 + (size_t)l * 6144, v26 + (size_t)l * 6144,
                                                scb + (1 + l) * 8192, shb + (1 + l) * 8192, 6144);
  bnfold_kernel<<<dim3(32),  blk, 0, stream>>>(g7, be7, m7, v7, scb + 6 * 8192, shb + 6 * 8192, 8192);
  bnfold_kernel<<<dim3(8),   blk, 0, stream>>>(g8, be8, m8, v8_, scb + 7 * 8192, shb + 7 * 8192, 2048);

  // L1: stat[4608,768] x w1[768,6144] -> xA
  gemm_bn_split<<<dim3(36, 48), blk, 0, stream>>>(stat, stat, stat, 768, 768, 768, 768, 768,
                                                  w1, 6144, 768, scb + 0 * 8192, shb + 0 * 8192, cb1, xA);
  // L2..L6: ping-pong 6144x6144 GEMMs; result lands in xB
  float* bufs[2] = { xA, xB };
  for (int l = 0; l < 5; ++l) {
    const float* src = bufs[l & 1];
    float* dst = bufs[(l + 1) & 1];
    gemm_bn_split<<<dim3(36, 48), blk, 0, stream>>>(src, src, src, 6144, 6144, 6144, 6144, 6144,
                                                    w26 + (size_t)l * 6144 * 6144, 6144, 6144,
                                                    scb + (1 + l) * 8192, shb + (1 + l) * 8192,
                                                    cb26 + (size_t)l * 6144, dst);
  }
  // L7: concat(x6=xB | x0 | stat), K=8192 -> x7 [4608,2048]
  gemm_bn_split<<<dim3(36, 16), blk, 0, stream>>>(xB, x0, stat, 6144, 7424, 6144, 1280, 768,
                                                  w7, 2048, 8192, scb + 6 * 8192, shb + 6 * 8192, cb7, x7);
  // L8: x7 x w8 -> x8 [4608,2048]
  gemm_bn_split<<<dim3(36, 16), blk, 0, stream>>>(x7, x7, x7, 2048, 2048, 2048, 2048, 2048,
                                                  w8, 2048, 2048, scb + 7 * 8192, shb + 7 * 8192, cb8, x8);
  // Global average pool -> out [32,2048]
  gap_kernel<<<dim3(8, 32), blk, 0, stream>>>(x8, (float*)d_out);
}
